// RNN_45853070852516
// MI455X (gfx1250) — compile-verified, matrix operand order checked by reading, not run on hardware
//
#include <hip/hip_runtime.h>
#include <hip/hip_bf16.h>

// ---------------------------------------------------------------------------
// tanh-RNN scan for MI455X (gfx1250, wave32).
//   B=512, S=4096, H=40.
//   One wave owns 16 batch rows and runs the whole 4096-step scan.
//   Per step:  D[16x16] = A[16x64(f16)] x Waug[64x16(f16)] + 0   (x4 N-tiles)
//   A = [h_{t-1}(K0..39), x_t(K40), 1.0(K41), pad]; Waug rows 40/41 carry
//   w_ih and (b_ih+b_hh); N-tile 3 column 0 carries fc_w (+fc_b in row 41),
//   so input projection, biases and the output FC all ride the same WMMAs.
// ---------------------------------------------------------------------------

typedef __attribute__((ext_vector_type(16))) _Float16 v16h;
typedef __attribute__((ext_vector_type(8)))  float    v8f;
typedef unsigned int uv4 __attribute__((ext_vector_type(4)));

union V16H {
    v16h     v;
    unsigned u[8];
    uv4      q[2];
};

__device__ __forceinline__ v8f wmma_f16(v16h a, v16h b, v8f c) {
    return __builtin_amdgcn_wmma_f32_16x16x32_f16(
        /*neg_a=*/false, a, /*neg_b=*/false, b,
        /*c_mod=*/(short)0, c, /*reuse_a=*/false, /*reuse_b=*/false);
}

__device__ __forceinline__ float fast_tanh(float x) {
#if __has_builtin(__builtin_amdgcn_tanhf)
    return __builtin_amdgcn_tanhf(x);            // v_tanh_f32 (TRANS, co-issues with WMMA)
#else
    // tanh(x) = 1 - 2/(exp(2x)+1), exp via v_exp_f32 (TRANS)
    float e = __builtin_amdgcn_exp2f(x * 2.885390081777927f); // 2*log2(e)
    return 1.0f - 2.0f / (e + 1.0f);
#endif
}

#define RNN_B 512
#define RNN_S 4096
#define RNN_H 40

__global__ __launch_bounds__(32) void rnn_scan_wmma(
    const float* __restrict__ x,      // [B,S,1]
    const float* __restrict__ hidden, // [1,B,H]
    const float* __restrict__ w_ih,   // [H,1]
    const float* __restrict__ w_hh,   // [H,H]
    const float* __restrict__ b_ih,   // [H]
    const float* __restrict__ b_hh,   // [H]
    const float* __restrict__ fc_w,   // [1,H]
    const float* __restrict__ fc_b,   // [1]
    float* __restrict__ out,          // [B,S,1]
    float* __restrict__ hlast)        // [B,H]
{
    constexpr int S = RNN_S, H = RNN_H;
    constexpr int KP = 64;                 // padded K (H + x + 1 + pad)

    __shared__ _Float16 hbuf[16 * KP];     // h staging, row-major [16][64] f16

    const int lane = threadIdx.x & 31;
    const int half = lane >> 4;            // 0: lanes 0-15, 1: lanes 16-31
    const int ln   = lane & 15;
    const int m0   = blockIdx.x * 16;      // first batch row of this tile

    // ---- Build W_aug in f16 B-layout, resident in VGPRs for the whole scan.
    // B-layout (32x16 f16): lane<16 -> N=lane, K runs [0..7],[16..23];
    //                       lane>=16 -> N=lane-16, K runs [8..15],[24..31].
    const float fcb = fc_b[0];
    v16h bm[4][2];
    for (int nt = 0; nt < 4; ++nt) {
        for (int kc = 0; kc < 2; ++kc) {
            v16h bb;
            #pragma unroll
            for (int j = 0; j < 16; ++j) {
                const int run = j >> 3, jj = j & 7;
                const int k = kc * 32 + run * 16 + half * 8 + jj;
                float v = 0.0f;
                if (nt < 3) {
                    const int n = ln + 16 * nt;          // output hidden index
                    if (n < H) {
                        if      (k < H)      v = w_hh[n * H + k]; // (h @ W_hh^T)
                        else if (k == H)     v = w_ih[n];         // x term
                        else if (k == H + 1) v = b_ih[n] + b_hh[n];
                    }
                } else if (ln == 0) {                    // FC tile, column 0 only
                    if      (k < H)      v = fc_w[k];
                    else if (k == H + 1) v = fcb;        // rides the A[K=41]=1 row
                }
                bb[j] = (_Float16)v;
            }
            bm[nt][kc] = bb;
        }
    }

    // ---- Init LDS with h0 (from `hidden`), zero-padded to 64 columns.
    for (int i = threadIdx.x; i < 16 * KP; i += 32) {
        const int row = i >> 6, col = i & 63;
        hbuf[i] = (_Float16)((col < H) ? hidden[(size_t)(m0 + row) * H + col] : 0.0f);
    }

    const float* xrow = x + (size_t)(m0 + ln) * S;   // hi lanes feed A[K=40]
    float xv = xrow[0];

    V16H a0, a1;
    v8f  hprev0 = {}, hprev1 = {}, hprev2 = {};      // last tanh(h) in f32 (C layout)

    #pragma clang loop unroll(disable)
    for (int t = 0; t < S; ++t) {
        // ---- Assemble A_t from LDS (transpose via wave-private LDS, in-order DS)
        {
            const _Float16* rp = hbuf + ln * KP;
            a0.q[0] = *(const uv4*)(rp +      half * 8);   // K 0..7 / 8..15
            a0.q[1] = *(const uv4*)(rp + 16 + half * 8);   // K 16..23 / 24..31
            a1.q[0] = *(const uv4*)(rp + 32 + half * 8);   // K 32..39 / 40..47
            a1.q[1] = *(const uv4*)(rp + 48 + half * 8);   // zero pad
            union { _Float16 h; unsigned short s; } cv; cv.h = (_Float16)xv;
            const unsigned px = (unsigned)cv.s | 0x3C000000u; // {x_t, 1.0h}
            if (half) a1.u[0] = px;                    // splice K=40,41 (hi lanes)
        }
        // Prefetch next x (independent of the recurrence chain)
        const int tn = (t + 1 < S) ? (t + 1) : (S - 1);
        const float xnext = xrow[tn];

        // ---- 8 WMMAs: 3 hidden N-tiles + 1 FC tile
        v8f c0 = {}, c1 = {}, c2 = {}, c3 = {};
        c0 = wmma_f16(a0.v, bm[0][0], c0);  c0 = wmma_f16(a1.v, bm[0][1], c0);
        c1 = wmma_f16(a0.v, bm[1][0], c1);  c1 = wmma_f16(a1.v, bm[1][1], c1);
        c2 = wmma_f16(a0.v, bm[2][0], c2);  c2 = wmma_f16(a1.v, bm[2][1], c2);
        c3 = wmma_f16(a0.v, bm[3][0], c3);  c3 = wmma_f16(a1.v, bm[3][1], c3);

        // ---- c3 column 0 = fc_w . h_{t-1} + fc_b  ->  out[., t-1]
        if (ln == 0 && t > 0) {
            const size_t base = (size_t)(m0 + half * 8) * S + (size_t)(t - 1);
            #pragma unroll
            for (int r = 0; r < 8; ++r) out[base + (size_t)r * S] = c3[r];
        }

        // ---- tanh (TRANS, co-issues with next wave's XDL) + f16 store to LDS
        _Float16* wp = hbuf + (half * 8) * KP + ln;  // row r+8*half, col ln+16*nt
        #pragma unroll
        for (int r = 0; r < 8; ++r) {
            const float h0v = fast_tanh(c0[r]);
            const float h1v = fast_tanh(c1[r]);
            const float h2v = fast_tanh(c2[r]);
            hprev0[r] = h0v; hprev1[r] = h1v; hprev2[r] = h2v;
            wp[r * KP + 0]  = (_Float16)h0v;
            wp[r * KP + 16] = (_Float16)h1v;
            wp[r * KP + 32] = (_Float16)h2v;   // cols 40..47 compute to tanh(0)=0
        }
        xv = xnext;
    }

    // ---- Epilogue: out[., S-1] needs h_{S-1} -> one more FC WMMA pair
    {
        V16H f0, f1;
        const _Float16* rp = hbuf + ln * KP;
        f0.q[0] = *(const uv4*)(rp +      half * 8);
        f0.q[1] = *(const uv4*)(rp + 16 + half * 8);
        f1.q[0] = *(const uv4*)(rp + 32 + half * 8);
        f1.q[1] = *(const uv4*)(rp + 48 + half * 8);
        if (half) f1.u[0] = 0x3C000000u;       // {0, 1.0h} -> keep fc_b term
        v8f c3 = {};
        c3 = wmma_f16(f0.v, bm[3][0], c3);
        c3 = wmma_f16(f1.v, bm[3][1], c3);
        if (ln == 0) {
            const size_t base = (size_t)(m0 + half * 8) * S + (size_t)(S - 1);
            #pragma unroll
            for (int r = 0; r < 8; ++r) out[base + (size_t)r * S] = c3[r];
        }
    }

    // ---- h_last [B,H] from the f32 tanh outputs of the final step
    #pragma unroll
    for (int nt = 0; nt < 3; ++nt) {
        const int n = ln + 16 * nt;
        if (n < H) {
            #pragma unroll
            for (int r = 0; r < 8; ++r) {
                const float hv = (nt == 0) ? hprev0[r] : (nt == 1) ? hprev1[r] : hprev2[r];
                hlast[(size_t)(m0 + r + half * 8) * H + n] = hv;
            }
        }
    }
}

extern "C" void kernel_launch(void* const* d_in, const int* in_sizes, int n_in,
                              void* d_out, int out_size, void* d_ws, size_t ws_size,
                              hipStream_t stream) {
    const float* x      = (const float*)d_in[0];
    const float* hidden = (const float*)d_in[1];
    const float* w_ih   = (const float*)d_in[2];
    const float* w_hh   = (const float*)d_in[3];
    const float* b_ih   = (const float*)d_in[4];
    const float* b_hh   = (const float*)d_in[5];
    const float* fc_w   = (const float*)d_in[6];
    const float* fc_b   = (const float*)d_in[7];

    float* out   = (float*)d_out;                       // [B,S,1] first
    float* hlast = out + (size_t)RNN_B * RNN_S;         // then [1,B,H]

    rnn_scan_wmma<<<RNN_B / 16, 32, 0, stream>>>(
        x, hidden, w_ih, w_hh, b_ih, b_hh, fc_w, fc_b, out, hlast);
}
